// Dpam_20409684590755
// MI455X (gfx1250) — matrix operand order repaired and checked by reading, not compiled, not run on hardware
//
#include <hip/hip_runtime.h>

// ---------------------------------------------------------------------------
// DPAM forward for MI455X (gfx1250, wave32, WMMA bf16 16x16x32).
// Pipeline:
//   xT   = bf16 transpose of x                    (B,4096,128)
//   S    = xT * xT^T  (gram, WMMA bf16)           (B,4096,4096) bf16
//   r,is = row max / 1/sum(exp) of S              (B,4096) f32
//   S   <- A^T in place (softmax + transpose, symmetric-pair tiles)
//   layer i<2:  u = bf16(W_i @ x_i)  (VALU conv); y = u @ A (WMMA, B-op = rows of A^T)
//   layer i>=2: y = W_i @ x_i
//   BN (training stats over B*N per channel) + ReLU after every layer
//   p = channel softmax; l[b] = ||I - p p^T||_F
// ---------------------------------------------------------------------------

typedef __attribute__((ext_vector_type(16))) __bf16 v16bf;
typedef __attribute__((ext_vector_type(8)))  __bf16 v8bf;
typedef __attribute__((ext_vector_type(8)))  float  v8f;

#define NPOS   4096
#define NBATCH 4
#define C0     128

static __device__ __forceinline__ v16bf ldfrag(const __bf16* p0, const __bf16* p1) {
  v8bf lo = *(const v8bf*)p0;
  v8bf hi = *(const v8bf*)p1;
  return __builtin_shufflevector(lo, hi, 0,1,2,3,4,5,6,7,8,9,10,11,12,13,14,15);
}

// ---- x (B,128,N) f32  ->  xT (B,N,128) bf16 (LDS-tiled transpose) ----------
__global__ __launch_bounds__(256) void k_cvt_xT(const float* __restrict__ x,
                                                __bf16* __restrict__ xT) {
  __shared__ __bf16 tl[64][33];
  const int b = blockIdx.z;
  const int n0 = blockIdx.x * 64, c0 = blockIdx.y * 32;
  const int t = threadIdx.x;
  {
    const int nn = t & 63, cs = t >> 6;
    #pragma unroll
    for (int st = 0; st < 8; st++) {
      const int cc = cs + st * 4;
      tl[nn][cc] = (__bf16)x[((size_t)b * C0 + c0 + cc) * NPOS + n0 + nn];
    }
  }
  __syncthreads();
  {
    const int cc = t & 31, ns = t >> 5;
    #pragma unroll
    for (int st = 0; st < 8; st++) {
      const int nn = ns + st * 8;
      xT[((size_t)b * NPOS + n0 + nn) * C0 + c0 + cc] = tl[nn][cc];
    }
  }
}

// ---- S = X^T X, bf16 out, WMMA. block = 128x128 tile, 8 waves of 64x32 -----
__global__ __launch_bounds__(256) void k_gram(const __bf16* __restrict__ xT,
                                              __bf16* __restrict__ S) {
  const int b = blockIdx.z;
  const int tid = threadIdx.x, lane = tid & 31, w = tid >> 5;
  const int wm = w >> 2, wn = w & 3;
  const __bf16* X = xT + (size_t)b * NPOS * C0;
  __bf16* Sb = S + (size_t)b * NPOS * NPOS;
  const int m0 = blockIdx.y * 128 + wm * 64;
  const int n0 = blockIdx.x * 128 + wn * 32;
  const int l15 = lane & 15;
  const int hiA = (lane < 16) ? 0 : 8;
  const int hiB = (lane < 16) ? 0 : 8;

  v8f acc[4][2] = {};
  #pragma unroll
  for (int kt = 0; kt < 4; kt++) {            // K = 128 channels, 4 x 32
    const int k0 = kt * 32;
    v16bf af[4], bf[2];
    #pragma unroll
    for (int i = 0; i < 4; i++) {
      const __bf16* pa = X + (size_t)(m0 + i * 16 + l15) * C0 + k0 + hiA;
      af[i] = ldfrag(pa, pa + 16);
    }
    #pragma unroll
    for (int j = 0; j < 2; j++) {
      const __bf16* pb = X + (size_t)(n0 + j * 16 + l15) * C0 + k0 + hiB;
      bf[j] = ldfrag(pb, pb + 16);
    }
    #pragma unroll
    for (int i = 0; i < 4; i++)
      #pragma unroll
      for (int j = 0; j < 2; j++)
        acc[i][j] = __builtin_amdgcn_wmma_f32_16x16x32_bf16(
            false, af[i], false, bf[j], (short)0, acc[i][j], false, false);
  }
  const int rbase = m0 + ((lane < 16) ? 0 : 8);
  #pragma unroll
  for (int i = 0; i < 4; i++)
    #pragma unroll
    for (int j = 0; j < 2; j++)
      #pragma unroll
      for (int v = 0; v < 8; v++)
        Sb[(size_t)(rbase + i * 16 + v) * NPOS + n0 + j * 16 + l15] =
            (__bf16)acc[i][j][v];
}

// ---- per-row softmax statistics: r = max, is = 1/sum(exp(s-max)) -----------
__global__ __launch_bounds__(256) void k_rowstats(const __bf16* __restrict__ S,
                                                  float* __restrict__ r,
                                                  float* __restrict__ is) {
  const int row = blockIdx.x;                 // b*4096 + n
  const __bf16* p = S + (size_t)row * NPOS;
  const int t = threadIdx.x;
  float v[16], m = -3.4e38f;
  #pragma unroll
  for (int i = 0; i < 16; i++) { v[i] = (float)p[t + i * 256]; m = fmaxf(m, v[i]); }
  __shared__ float red[256];
  red[t] = m; __syncthreads();
  for (int s = 128; s > 0; s >>= 1) { if (t < s) red[t] = fmaxf(red[t], red[t + s]); __syncthreads(); }
  m = red[0]; __syncthreads();
  float ssum = 0.f;
  #pragma unroll
  for (int i = 0; i < 16; i++) ssum += __expf(v[i] - m);
  red[t] = ssum; __syncthreads();
  for (int s = 128; s > 0; s >>= 1) { if (t < s) red[t] += red[t + s]; __syncthreads(); }
  if (t == 0) { r[row] = m; is[row] = 1.0f / red[0]; }
}

// ---- S <- A^T in place. Block handles symmetric tile pair (it,jt), jt>=it --
__global__ __launch_bounds__(256) void k_softAT(__bf16* __restrict__ S,
                                                const float* __restrict__ r,
                                                const float* __restrict__ is) {
  const int jt = blockIdx.x, it = blockIdx.y, b = blockIdx.z;
  if (jt < it) return;
  __bf16* Sb = S + (size_t)b * NPOS * NPOS;
  const float* rb = r + (size_t)b * NPOS;
  const float* sb = is + (size_t)b * NPOS;
  __shared__ __bf16 A0[64][65];
  __shared__ __bf16 A1[64][65];
  const int t = threadIdx.x, c = t & 63, r4 = t >> 6;
  const int ni = it * 64, nj = jt * 64;
  #pragma unroll
  for (int st = 0; st < 16; st++) {           // A0[x][y] = f(S[ni+x][nj+y])
    const int rr = r4 + st * 4;
    const float sv = (float)Sb[(size_t)(ni + rr) * NPOS + nj + c];
    A0[rr][c] = (__bf16)(__expf(sv - rb[ni + rr]) * sb[ni + rr]);
  }
  if (jt > it) {
    #pragma unroll
    for (int st = 0; st < 16; st++) {         // A1[x][y] = f(S[nj+x][ni+y])
      const int rr = r4 + st * 4;
      const float sv = (float)Sb[(size_t)(nj + rr) * NPOS + ni + c];
      A1[rr][c] = (__bf16)(__expf(sv - rb[nj + rr]) * sb[nj + rr]);
    }
  }
  __syncthreads();
  #pragma unroll
  for (int st = 0; st < 16; st++) {           // AT[nj+rr][ni+c] = A0[c][rr]
    const int rr = r4 + st * 4;
    Sb[(size_t)(nj + rr) * NPOS + ni + c] = A0[c][rr];
  }
  if (jt > it) {
    #pragma unroll
    for (int st = 0; st < 16; st++) {         // AT[ni+rr][nj+c] = A1[c][rr]
      const int rr = r4 + st * 4;
      Sb[(size_t)(ni + rr) * NPOS + nj + c] = A1[c][rr];
    }
  }
}

// ---- 1x1 conv: out[b,co,n] = sum_ci W[co,ci] * x[b,ci,n] -------------------
template <int CIN, int COUT, bool BF16OUT>
__global__ __launch_bounds__(256) void k_conv(const float* __restrict__ x,
                                              const float* __restrict__ w,
                                              void* __restrict__ outv) {
  const int b = blockIdx.y;
  const int n = blockIdx.x * 256 + threadIdx.x;
  const float* xb = x + (size_t)b * CIN * NPOS + n;
  float acc[COUT];
  #pragma unroll
  for (int co = 0; co < COUT; co++) acc[co] = 0.f;
  for (int ci = 0; ci < CIN; ci++) {
    const float xv = xb[(size_t)ci * NPOS];
    #pragma unroll
    for (int co = 0; co < COUT; co++) acc[co] = fmaf(w[co * CIN + ci], xv, acc[co]);
  }
  if constexpr (BF16OUT) {
    __bf16* o = (__bf16*)outv;
    #pragma unroll
    for (int co = 0; co < COUT; co++)
      o[((size_t)b * COUT + co) * NPOS + n] = (__bf16)acc[co];
  } else {
    float* o = (float*)outv;
    #pragma unroll
    for (int co = 0; co < COUT; co++)
      o[((size_t)b * COUT + co) * NPOS + n] = acc[co];
  }
}

// ---- y = u @ A  (M=128, N=4096, K=4096). A-op rows of u, B-op rows of A^T --
__global__ __launch_bounds__(256) void k_mm(const __bf16* __restrict__ u,
                                            const __bf16* __restrict__ AT,
                                            float* __restrict__ y) {
  const int b = blockIdx.y;
  const int tid = threadIdx.x, lane = tid & 31, w = tid >> 5;
  const int wm = w >> 2, wn = w & 3;
  const __bf16* U = u + (size_t)b * C0 * NPOS;
  const __bf16* T = AT + (size_t)b * NPOS * NPOS;
  float* Y = y + (size_t)b * C0 * NPOS;
  const int m0 = wm * 64;                     // M = 128 covered by one block
  const int n0 = blockIdx.x * 128 + wn * 32;
  const int l15 = lane & 15;
  const int hiA = (lane < 16) ? 0 : 8;
  const int hiB = (lane < 16) ? 0 : 8;

  v8f acc[4][2] = {};
  #pragma unroll 2
  for (int kt = 0; kt < 128; kt++) {          // K = 4096, 128 x 32
    const int k0 = kt * 32;
    v16bf af[4], bf[2];
    #pragma unroll
    for (int i = 0; i < 4; i++) {
      const __bf16* pa = U + (size_t)(m0 + i * 16 + l15) * NPOS + k0 + hiA;
      af[i] = ldfrag(pa, pa + 16);
    }
    #pragma unroll
    for (int j = 0; j < 2; j++) {
      const __bf16* pb = T + (size_t)(n0 + j * 16 + l15) * NPOS + k0 + hiB;
      __builtin_prefetch(pb + 2048, 0, 1);    // stream-ahead on the 128MB A^T
      bf[j] = ldfrag(pb, pb + 16);
    }
    #pragma unroll
    for (int i = 0; i < 4; i++)
      #pragma unroll
      for (int j = 0; j < 2; j++)
        acc[i][j] = __builtin_amdgcn_wmma_f32_16x16x32_bf16(
            false, af[i], false, bf[j], (short)0, acc[i][j], false, false);
  }
  const int rbase = m0 + ((lane < 16) ? 0 : 8);
  #pragma unroll
  for (int i = 0; i < 4; i++)
    #pragma unroll
    for (int j = 0; j < 2; j++)
      #pragma unroll
      for (int v = 0; v < 8; v++)
        Y[(size_t)(rbase + i * 16 + v) * NPOS + n0 + j * 16 + l15] = acc[i][j][v];
}

// ---- BN training-mode stats per channel over B*N ---------------------------
__global__ __launch_bounds__(256) void k_bnstats(const float* __restrict__ y, int C,
                                                 float* __restrict__ mu,
                                                 float* __restrict__ rv) {
  const int c = blockIdx.x, t = threadIdx.x;
  float s = 0.f, s2 = 0.f;
  for (int b = 0; b < NBATCH; b++) {
    const float* p = y + ((size_t)b * C + c) * NPOS;
    for (int n = t; n < NPOS; n += 256) { const float v = p[n]; s += v; s2 = fmaf(v, v, s2); }
  }
  __shared__ float r1[256], r2[256];
  r1[t] = s; r2[t] = s2; __syncthreads();
  for (int st = 128; st > 0; st >>= 1) {
    if (t < st) { r1[t] += r1[t + st]; r2[t] += r2[t + st]; }
    __syncthreads();
  }
  if (t == 0) {
    const float inv = 1.0f / (float)(NBATCH * NPOS);
    const float m = r1[0] * inv;
    const float var = r2[0] * inv - m * m;
    mu[c] = m; rv[c] = rsqrtf(var + 1e-5f);
  }
}

__global__ void k_bnapply(const float* __restrict__ y, int C,
                          const float* __restrict__ mu, const float* __restrict__ rv,
                          const float* __restrict__ g, const float* __restrict__ bb,
                          float* __restrict__ xo) {
  const size_t idx = (size_t)blockIdx.x * 256 + threadIdx.x;
  const int c = (int)((idx >> 12) & (size_t)(C - 1));
  const float v = (y[idx] - mu[c]) * rv[c] * g[c] + bb[c];
  xo[idx] = fmaxf(v, 0.f);
}

// ---- channel softmax (32 ch) ----------------------------------------------
__global__ __launch_bounds__(256) void k_chsm(const float* __restrict__ x4,
                                              float* __restrict__ p) {
  const int idx = blockIdx.x * 256 + threadIdx.x;   // B*N = 16384
  const int b = idx >> 12, n = idx & 4095;
  const float* px = x4 + (size_t)b * 32 * NPOS + n;
  float v[32], m = -3.4e38f;
  #pragma unroll
  for (int c = 0; c < 32; c++) { v[c] = px[(size_t)c * NPOS]; m = fmaxf(m, v[c]); }
  float s = 0.f;
  #pragma unroll
  for (int c = 0; c < 32; c++) { v[c] = __expf(v[c] - m); s += v[c]; }
  const float inv = 1.0f / s;
  float* po = p + (size_t)b * 32 * NPOS + n;
  #pragma unroll
  for (int c = 0; c < 32; c++) po[(size_t)c * NPOS] = v[c] * inv;
}

// ---- l[b] = || I - p p^T ||_F ---------------------------------------------
__global__ __launch_bounds__(256) void k_fro(const float* __restrict__ p,
                                             float* __restrict__ l) {
  const int b = blockIdx.x, t = threadIdx.x;
  const float* pb = p + (size_t)b * 32 * NPOS;
  float local = 0.f;
  for (int pr = t; pr < 1024; pr += 256) {
    const int c = pr >> 5, d = pr & 31;
    const float* pc = pb + (size_t)c * NPOS;
    const float* pd = pb + (size_t)d * NPOS;
    float s = 0.f;
    for (int n = 0; n < NPOS; n++) s = fmaf(pc[n], pd[n], s);
    const float diff = ((c == d) ? 1.0f : 0.0f) - s;
    local += diff * diff;
  }
  __shared__ float red[256];
  red[t] = local; __syncthreads();
  for (int s = 128; s > 0; s >>= 1) { if (t < s) red[t] += red[t + s]; __syncthreads(); }
  if (t == 0) l[b] = sqrtf(red[0]);
}

// ---------------------------------------------------------------------------
extern "C" void kernel_launch(void* const* d_in, const int* in_sizes, int n_in,
                              void* d_out, int out_size, void* d_ws, size_t ws_size,
                              hipStream_t stream) {
  // dict order: x, w0,g0,b0, w1,g1,b1, w2,g2,b2, w3,g3,b3
  const float* x  = (const float*)d_in[0];
  const float* w0 = (const float*)d_in[1];
  const float* g0 = (const float*)d_in[2];
  const float* b0 = (const float*)d_in[3];
  const float* w1 = (const float*)d_in[4];
  const float* g1 = (const float*)d_in[5];
  const float* b1 = (const float*)d_in[6];
  const float* w2 = (const float*)d_in[7];
  const float* g2 = (const float*)d_in[8];
  const float* b2 = (const float*)d_in[9];
  const float* w3 = (const float*)d_in[10];
  const float* g3 = (const float*)d_in[11];
  const float* b3 = (const float*)d_in[12];
  float* out = (float*)d_out;                       // p (B*32*4096) then l (B)

  char* wp = (char*)d_ws;
  auto alloc = [&](size_t bytes) -> void* {
    void* p = (void*)wp; wp += (bytes + 255) & ~(size_t)255; return p;
  };
  __bf16* xT = (__bf16*)alloc((size_t)NBATCH * NPOS * C0 * 2);          // 4 MB
  __bf16* S  = (__bf16*)alloc((size_t)NBATCH * NPOS * NPOS * 2);        // 128 MB
  float*  rr = (float*)alloc((size_t)NBATCH * NPOS * 4);
  float*  is = (float*)alloc((size_t)NBATCH * NPOS * 4);
  __bf16* u  = (__bf16*)alloc((size_t)NBATCH * C0 * NPOS * 2);          // 4 MB
  float*  y  = (float*)alloc((size_t)NBATCH * C0 * NPOS * 4);           // 8 MB
  float*  y2 = (float*)alloc((size_t)NBATCH * 64 * NPOS * 4);           // 4 MB
  float*  y3 = (float*)alloc((size_t)NBATCH * 32 * NPOS * 4);           // 2 MB
  float*  mu = (float*)alloc(C0 * 4);
  float*  rv = (float*)alloc(C0 * 4);

  // attention matrix
  k_cvt_xT<<<dim3(64, 4, NBATCH), 256, 0, stream>>>(x, xT);
  k_gram<<<dim3(32, 32, NBATCH), 256, 0, stream>>>(xT, S);
  k_rowstats<<<NBATCH * NPOS, 256, 0, stream>>>(S, rr, is);
  k_softAT<<<dim3(64, 64, NBATCH), 256, 0, stream>>>(S, rr, is);

  // layer 0: y = (W0 x) A ; BN ; ReLU   (in-place BN into y)
  k_conv<128, 128, true><<<dim3(16, NBATCH), 256, 0, stream>>>(x, w0, u);
  k_mm<<<dim3(32, NBATCH), 256, 0, stream>>>(u, S, y);
  k_bnstats<<<128, 256, 0, stream>>>(y, 128, mu, rv);
  k_bnapply<<<8192, 256, 0, stream>>>(y, 128, mu, rv, g0, b0, y);

  // layer 1
  k_conv<128, 128, true><<<dim3(16, NBATCH), 256, 0, stream>>>(y, w1, u);
  k_mm<<<dim3(32, NBATCH), 256, 0, stream>>>(u, S, y);
  k_bnstats<<<128, 256, 0, stream>>>(y, 128, mu, rv);
  k_bnapply<<<8192, 256, 0, stream>>>(y, 128, mu, rv, g1, b1, y);

  // layer 2 (no attention)
  k_conv<128, 64, false><<<dim3(16, NBATCH), 256, 0, stream>>>(y, w2, y2);
  k_bnstats<<<64, 256, 0, stream>>>(y2, 64, mu, rv);
  k_bnapply<<<4096, 256, 0, stream>>>(y2, 64, mu, rv, g2, b2, y2);

  // layer 3
  k_conv<64, 32, false><<<dim3(16, NBATCH), 256, 0, stream>>>(y2, w3, y3);
  k_bnstats<<<32, 256, 0, stream>>>(y3, 32, mu, rv);
  k_bnapply<<<2048, 256, 0, stream>>>(y3, 32, mu, rv, g3, b3, y3);

  // outputs
  k_chsm<<<64, 256, 0, stream>>>(y3, out);
  k_fro<<<NBATCH, 256, 0, stream>>>(out, out + (size_t)NBATCH * 32 * NPOS);
}
